// GATLayer_25280177504780
// MI455X (gfx1250) — matrix operand order repaired and checked by reading, not compiled, not run on hardware
//
#include <hip/hip_runtime.h>

// Problem constants (match reference)
#define BATCH    4
#define NNODES   2048
#define IN_DIM   256
#define OUT_DIMX 256
#define HEADS    4
#define PHD      64
#define MTOT     (BATCH * NNODES)   // 8192 rows total
#define ADJW     (NNODES / 32)      // 64 bitmask words per row
#define VPITCH   72                 // LDS row pitch in halves (bank-conflict-free)

typedef __attribute__((ext_vector_type(16))) _Float16 v16h;
typedef __attribute__((ext_vector_type(8)))  _Float16 h8;
typedef __attribute__((ext_vector_type(8)))  float    v8f;
typedef __attribute__((ext_vector_type(4)))  float    v4f;
typedef __attribute__((ext_vector_type(4)))  int      v4i;
typedef __attribute__((ext_vector_type(2)))  unsigned v2u;

static __device__ __forceinline__ v8f wmma_f16(v16h a, v16h b, v8f c) {
  // 8 args: (neg_a, A, neg_b, B, c_mod, C, reuse_a, reuse_b)
  return __builtin_amdgcn_wmma_f32_16x16x32_f16(false, a, false, b, (short)0, c,
                                                false, false);
}

// ---------------------------------------------------------------------------
// Kernel 0: pack int32 adjacency into a bitmask [2048][64 words].
// Cuts flash-kernel mask traffic 32x (256 MB -> 8 MB of L2 reads).
// ---------------------------------------------------------------------------
__global__ __launch_bounds__(256) void gat_pack_adj_kernel(
    const int* __restrict__ adj, unsigned* __restrict__ bits) {
  const int idx = blockIdx.x * 256 + threadIdx.x;   // 131072 words
  const int* p = adj + (size_t)idx * 32;            // 2048 cols = 64 words/row
  unsigned w = 0;
#pragma unroll
  for (int k = 0; k < 32; k += 4) {
    const v4i a = *(const v4i*)(p + k);
    w |= (a[0] ? 1u : 0u) << (k + 0);
    w |= (a[1] ? 1u : 0u) << (k + 1);
    w |= (a[2] ? 1u : 0u) << (k + 2);
    w |= (a[3] ? 1u : 0u) << (k + 3);
  }
  bits[idx] = w;
}

// ---------------------------------------------------------------------------
// Kernel 1: projection GEMM  proj = X @ W^T, output stored head-transposed as
// f16: projT[(b*H + h)*PHD + d][n]  (so aggregation B-fragments are contiguous)
// One wave per 16x16 output tile, K=256 in 8 WMMA steps of 32.
// 16-bit A layout: lane m=lane&15, element e -> k = (e<8 ? e : e+8) + 8*(lane>>4)
// 16-bit B layout: lane n=lane&15, element e -> k = (e<8 ? e : e+8) + 8*(lane>>4)
// ---------------------------------------------------------------------------
__global__ __launch_bounds__(128) void gat_proj_kernel(
    const float* __restrict__ x, const float* __restrict__ W,
    _Float16* __restrict__ projT) {
  const int lane = threadIdx.x & 31;
  const int wave = threadIdx.x >> 5;
  const int tile = blockIdx.x * 4 + wave;   // 8192 tiles total
  const int tm = tile >> 4;                 // 512 row tiles
  const int to = tile & 15;                 // 16 col tiles
  const int m0 = tm << 4, o0 = to << 4;
  const int l15 = lane & 15, hi = lane >> 4;

  v8f c = {};
  const float* xrow = x + (size_t)(m0 + l15) * IN_DIM;
  const float* wrow = W + (size_t)(o0 + l15) * IN_DIM;

#pragma unroll
  for (int k0 = 0; k0 < IN_DIM; k0 += 32) {
    // A fragment: k = (e<8 ? e : e+8) + 8*hi  -> two contiguous 8-float runs
    const float* ax = xrow + k0 + 8 * hi;
    v4f x0 = *(const v4f*)(ax);
    v4f x1 = *(const v4f*)(ax + 4);
    v4f x2 = *(const v4f*)(ax + 16);
    v4f x3 = *(const v4f*)(ax + 20);
    // B fragment: k = (e<8 ? e : e+8) + 8*hi -> two contiguous 8-float runs
    const float* bx = wrow + k0 + 8 * hi;
    v4f w0 = *(const v4f*)(bx);
    v4f w1 = *(const v4f*)(bx + 4);
    v4f w2 = *(const v4f*)(bx + 16);
    v4f w3 = *(const v4f*)(bx + 20);

    v16h a, b;
#pragma unroll
    for (int e = 0; e < 4; ++e) {
      a[e]      = (_Float16)x0[e];
      a[4 + e]  = (_Float16)x1[e];
      a[8 + e]  = (_Float16)x2[e];
      a[12 + e] = (_Float16)x3[e];
      b[e]      = (_Float16)w0[e];
      b[4 + e]  = (_Float16)w1[e];
      b[8 + e]  = (_Float16)w2[e];
      b[12 + e] = (_Float16)w3[e];
    }
    c = wmma_f16(a, b, c);
  }

  // C layout: lane col n=l15; VGPR r holds row m = r + 8*hi
#pragma unroll
  for (int r = 0; r < 8; ++r) {
    const int row  = m0 + r + 8 * hi;      // global row in [0,8192)
    const int bidx = row >> 11;            // batch
    const int n    = row & (NNODES - 1);   // node
    const int o    = o0 + l15;             // output feature
    const int h    = o >> 6, d = o & 63;
    projT[((size_t)((bidx * HEADS + h) * PHD + d)) * NNODES + n] = (_Float16)c[r];
  }
}

// ---------------------------------------------------------------------------
// Kernel 2: per-node attention logits  src[b,h,n] = proj . attn_src[h]
// Reads head-transposed proj -> fully coalesced across n.
// ---------------------------------------------------------------------------
__global__ __launch_bounds__(256) void gat_srcdst_kernel(
    const _Float16* __restrict__ projT,
    const float* __restrict__ attn_src, const float* __restrict__ attn_dst,
    float* __restrict__ srcv, float* __restrict__ dstv) {
  const int idx = blockIdx.x * blockDim.x + threadIdx.x;   // 32768 = B*H*N
  const int bh = idx >> 11;
  const int n  = idx & (NNODES - 1);
  const int h  = bh & (HEADS - 1);
  const _Float16* p = projT + (size_t)bh * PHD * NNODES + n;
  float s = 0.f, t = 0.f;
#pragma unroll
  for (int d = 0; d < PHD; ++d) {
    const float v = (float)p[(size_t)d * NNODES];
    s += v * attn_src[h * PHD + d];
    t += v * attn_dst[h * PHD + d];
  }
  srcv[idx] = s;
  dstv[idx] = t;
}

// ---------------------------------------------------------------------------
// Kernel 3: fused flash-style masked softmax + aggregation.
// Block = 4 waves = 4 row-tiles of the SAME (b,h); they iterate j-blocks in
// lockstep and share each 64x64 f16 V tile through double-buffered LDS
// (cuts the V stream from L2 4x). Column blocks of 64 = two WMMA K-steps per
// online-softmax update. Adjacency from packed bitmask (one b64/lane/iter).
// ---------------------------------------------------------------------------
__global__ __launch_bounds__(128) void gat_flash_kernel(
    const _Float16* __restrict__ projT,
    const float* __restrict__ srcv, const float* __restrict__ dstv,
    const unsigned* __restrict__ adjBits, const float* __restrict__ bias,
    float* __restrict__ out) {
  __shared__ _Float16 Vs[2][PHD * VPITCH];   // 2 x 9 KB, pitch 36 DWORDs

  const int lane = threadIdx.x & 31;
  const int wave = threadIdx.x >> 5;
  const int tile = blockIdx.x * 4 + wave;   // 2048 tiles; block-uniform bh
  const int bh = tile >> 7;                 // 16 (b,h) pairs
  const int it = tile & 127;
  const int i0 = it << 4;
  const int b = bh >> 2, h = bh & 3;
  const int l15 = lane & 15, hi = lane >> 4;
  const float NEGINF = -__builtin_inff();

  const float src_i = srcv[bh * NNODES + i0 + l15];          // row of this lane
  const float* dstB = dstv + bh * NNODES;
  const unsigned* adjRow = adjBits + (size_t)(i0 + l15) * ADJW;
  const _Float16* pbase = projT + (size_t)bh * PHD * NNODES; // block-uniform

  // cooperative fill of one 64(j) x 64(d) f16 tile: 128 threads x 4 chunks
  auto fillV = [&](int buf, int jbase) {
#pragma unroll
    for (int i = 0; i < 4; ++i) {
      const int c  = threadIdx.x + 128 * i;  // 0..511 chunks of 16B
      const int d  = c >> 3;
      const int jc = (c & 7) * 8;
      const h8 v = *(const h8*)(pbase + (size_t)d * NNODES + jbase + jc);
      *(h8*)(&Vs[buf][d * VPITCH + jc]) = v;
    }
  };

  v8f acc0 = {}, acc1 = {}, acc2 = {}, acc3 = {};
  float m_run = NEGINF;   // running max for row l15 (duplicated in lane halves)
  float l_run = 0.f;      // running sum for row l15

  fillV(0, 0);
  int p = 0;
  for (int j0 = 0; j0 < NNODES; j0 += 64) {
    __syncthreads();                         // Vs[p] ready for whole block
    if (j0 + 64 < NNODES) fillV(p ^ 1, j0 + 64);   // prefetch next tile

    // adjacency bits: words (j0/32, j0/32+1), 8B aligned -> one b64 load
    const v2u wAB = *(const v2u*)(adjRow + (j0 >> 5));

    // ---- scores in A-fragment layout over two 32-col sub-blocks
    // sub-block s, element e: j = j0 + 32*s + kloc(e),
    // kloc(e) = (e<8 ? e : e+8) + 8*hi  -> contiguous 8-float runs for dst
    const float* dp = dstB + j0 + 8 * hi;
    v4f d0 = *(const v4f*)(dp),      d1 = *(const v4f*)(dp + 4);
    v4f d2 = *(const v4f*)(dp + 16), d3 = *(const v4f*)(dp + 20);
    v4f d4 = *(const v4f*)(dp + 32), d5 = *(const v4f*)(dp + 36);
    v4f d6 = *(const v4f*)(dp + 48), d7 = *(const v4f*)(dp + 52);

    float sv[32];
    float lmax = NEGINF;
#pragma unroll
    for (int s = 0; s < 2; ++s) {
      const unsigned w = wAB[s];
#pragma unroll
      for (int e = 0; e < 16; ++e) {
        const int kloc = ((e < 8) ? e : e + 8) + 8 * hi;   // 0..31
        float dj;
        if (s == 0)
          dj = (e < 4) ? d0[e] : (e < 8) ? d1[e - 4]
               : (e < 12) ? d2[e - 8] : d3[e - 12];
        else
          dj = (e < 4) ? d4[e] : (e < 8) ? d5[e - 4]
               : (e < 12) ? d6[e - 8] : d7[e - 12];
        float sc = src_i + dj;
        sc = (sc >= 0.f) ? sc : 0.2f * sc;                 // leaky_relu(0.2)
        sc = ((w >> kloc) & 1u) ? sc : NEGINF;             // adjacency mask
        sv[s * 16 + e] = sc;
        lmax = fmaxf(lmax, sc);
      }
    }
    // row i lives in lanes {i, i+16}: combine halves
    const float rmax  = fmaxf(lmax, __shfl_xor(lmax, 16, 32));
    const float m_new = fmaxf(m_run, rmax);
    const float alpha = (m_run == NEGINF) ? 0.f : __expf(m_run - m_new);

    float psum = 0.f;
    v16h pa0, pa1;
#pragma unroll
    for (int e = 0; e < 16; ++e) {
      const float p0 = (sv[e] == NEGINF) ? 0.f : __expf(sv[e] - m_new);
      const float p1 = (sv[16 + e] == NEGINF) ? 0.f : __expf(sv[16 + e] - m_new);
      psum += p0 + p1;
      pa0[e] = (_Float16)p0;
      pa1[e] = (_Float16)p1;
    }
    l_run = l_run * alpha + (psum + __shfl_xor(psum, 16, 32));
    m_run = m_new;

    // rescale accumulators: C-row of VGPR r is (r + 8*hi); alpha sits in lane=row
#pragma unroll
    for (int r = 0; r < 8; ++r) {
      const float ar = __shfl(alpha, r + 8 * hi, 32);
      acc0[r] *= ar; acc1[r] *= ar; acc2[r] *= ar; acc3[r] *= ar;
    }

    // ---- V fragments from LDS: B[k][n] with k = (e<8 ? e : e+8) + 8*hi, n = l15
    // LDS offset (d*VPITCH + 32*s + 8*hi): two 8-half runs 16 apart per fragment
#pragma unroll
    for (int t = 0; t < 4; ++t) {
      const _Float16* pb = &Vs[p][(t * 16 + l15) * VPITCH + 8 * hi];
#pragma unroll
      for (int s = 0; s < 2; ++s) {
        h8 q0 = *(const h8*)(pb + 32 * s);
        h8 q1 = *(const h8*)(pb + 32 * s + 16);
        v16h vb;
#pragma unroll
        for (int e = 0; e < 8; ++e) { vb[e] = q0[e]; vb[8 + e] = q1[e]; }
        const v16h& pa = s ? pa1 : pa0;
        if (t == 0) acc0 = wmma_f16(pa, vb, acc0);
        if (t == 1) acc1 = wmma_f16(pa, vb, acc1);
        if (t == 2) acc2 = wmma_f16(pa, vb, acc2);
        if (t == 3) acc3 = wmma_f16(pa, vb, acc3);
      }
    }
    p ^= 1;
  }

  // ---- normalize and write out (+bias)
  const float inv = 1.f / l_run;  // row l15 (self-loops guarantee l_run > 0)
  const float* bp = bias + h * PHD;
  const float b0 = bp[l15], b1 = bp[16 + l15], b2 = bp[32 + l15], b3 = bp[48 + l15];
#pragma unroll
  for (int r = 0; r < 8; ++r) {
    const int rr = r + 8 * hi;
    const float w = __shfl(inv, rr, 32);
    const int row = i0 + rr;
    float* op = out + ((size_t)(b * NNODES + row)) * OUT_DIMX + h * PHD;
    op[l15]      = acc0[r] * w + b0;
    op[16 + l15] = acc1[r] * w + b1;
    op[32 + l15] = acc2[r] * w + b2;
    op[48 + l15] = acc3[r] * w + b3;
  }
}

// ---------------------------------------------------------------------------
// Host-side launcher
// ---------------------------------------------------------------------------
extern "C" void kernel_launch(void* const* d_in, const int* in_sizes, int n_in,
                              void* d_out, int out_size, void* d_ws, size_t ws_size,
                              hipStream_t stream) {
  const float* x        = (const float*)d_in[0];  // [4,2048,256]
  const int*   adjacency= (const int*)  d_in[1];  // [2048,2048]
  const float* W        = (const float*)d_in[2];  // [256,256]
  const float* attn_src = (const float*)d_in[3];  // [4,64]
  const float* attn_dst = (const float*)d_in[4];  // [4,64]
  const float* bias     = (const float*)d_in[5];  // [256]
  float* out = (float*)d_out;

  char* ws = (char*)d_ws;
  _Float16* projT = (_Float16*)ws;                                   // 4 MB
  float* srcv = (float*)(ws + (size_t)MTOT * OUT_DIMX * sizeof(_Float16));
  float* dstv = srcv + BATCH * HEADS * NNODES;                       // 128 KB each
  unsigned* adjBits = (unsigned*)(dstv + BATCH * HEADS * NNODES);    // 512 KB

  // 131072 bitmask words
  gat_pack_adj_kernel<<<512, 256, 0, stream>>>(adjacency, adjBits);
  // 8192 16x16 tiles / 4 waves per block
  gat_proj_kernel<<<2048, 128, 0, stream>>>(x, W, projT);
  // 32768 (b,h,n) logits
  gat_srcdst_kernel<<<128, 256, 0, stream>>>(projT, attn_src, attn_dst, srcv, dstv);
  // 2048 row tiles / 4 waves per block (block shares V tiles via LDS)
  gat_flash_kernel<<<512, 128, 0, stream>>>(projT, srcv, dstv, adjBits, bias, out);

  (void)in_sizes; (void)n_in; (void)out_size; (void)ws_size;
}
